// Decoder_18975165513807
// MI455X (gfx1250) — matrix operand order repaired and checked by reading, not compiled, not run on hardware
//
#include <hip/hip_runtime.h>
#include <hip/hip_bf16.h>

typedef __attribute__((ext_vector_type(16))) _Float16 v16h;
typedef __attribute__((ext_vector_type(8)))  _Float16 v8h;
typedef __attribute__((ext_vector_type(8)))  float    v8f;

#define B_  32
#define S_  256
#define NU_ 4
#define ML_ 9
#define H_  768
#define H3_ 2304
#define V_  30522
#define NTV 1908           // ceil(V/16)
#define NSTEP (NU_*ML_)

// ---------------------------------------------------------------------------
// M-blocked WMMA inner product over K=768: one wave computes a 32x16 tile
// (both 16-row M-tiles) sharing each B fragment.
// ap0/ap1: A + row*H + (lane>>4)*8 for m0=0 and m0=16 (A is MxK row-major, lda=H)
// bp:      B + col*H + (lane>>4)*8                   (B is NxK row-major, ldb=H)
// ---------------------------------------------------------------------------
__device__ __forceinline__ void wmma_k768_m32(const _Float16* __restrict__ ap0,
                                              const _Float16* __restrict__ ap1,
                                              const _Float16* __restrict__ bp,
                                              v8f& c0, v8f& c1) {
    c0 = (v8f){};
    c1 = (v8f){};
#pragma unroll 4
    for (int k0 = 0; k0 < H_; k0 += 32) {
        v8h b0 = *(const v8h*)(bp + k0);
        v8h b1 = *(const v8h*)(bp + k0 + 16);
        v16h b = __builtin_shufflevector(b0, b1, 0,1,2,3,4,5,6,7,8,9,10,11,12,13,14,15);

        v8h a00 = *(const v8h*)(ap0 + k0);
        v8h a01 = *(const v8h*)(ap0 + k0 + 16);
        v16h a0 = __builtin_shufflevector(a00, a01, 0,1,2,3,4,5,6,7,8,9,10,11,12,13,14,15);
        c0 = __builtin_amdgcn_wmma_f32_16x16x32_f16(false, a0, false, b, (short)0, c0,
                                                    false, false);
        v8h a10 = *(const v8h*)(ap1 + k0);
        v8h a11 = *(const v8h*)(ap1 + k0 + 16);
        v16h a1 = __builtin_shufflevector(a10, a11, 0,1,2,3,4,5,6,7,8,9,10,11,12,13,14,15);
        c1 = __builtin_amdgcn_wmma_f32_16x16x32_f16(false, a1, false, b, (short)0, c1,
                                                    false, false);
    }
}

// ---------------------------------------------------------------------------
// f32 -> f16 conversion
// ---------------------------------------------------------------------------
__global__ void cvt_f16_kernel(const float* __restrict__ src, _Float16* __restrict__ dst,
                               long n) {
    long i = (long)blockIdx.x * blockDim.x + threadIdx.x;
    if (i < n) dst[i] = (_Float16)src[i];
}

// ---------------------------------------------------------------------------
// Per-step prep: build w (decoder_input at t==0, else embed[teacher]), convert
// w and h to f16, zero softmax accumulator.  grid=B_, block=256
// ---------------------------------------------------------------------------
__global__ void prep_kernel(const float* __restrict__ dec_in,
                            const int* __restrict__ teacher,
                            const float* __restrict__ embed,
                            const float* __restrict__ h32,
                            _Float16* __restrict__ h16,
                            float* __restrict__ w32,
                            _Float16* __restrict__ w16,
                            float* __restrict__ sumExp,
                            int u, int t) {
    int b = blockIdx.x, tid = threadIdx.x;
    const float* src;
    if (t == 0) {
        src = dec_in + ((size_t)b * NU_ + u) * H_;
    } else {
        int tok = teacher[((size_t)b * NU_ + u) * ML_ + (t - 1)];
        src = embed + (size_t)tok * H_;
    }
#pragma unroll
    for (int j = 0; j < 3; ++j) {
        int k = tid + j * 256;
        float v = src[k];
        w32[b * H_ + k] = v;
        w16[b * H_ + k] = (_Float16)v;
        h16[b * H_ + k] = (_Float16)h32[b * H_ + k];
    }
    if (tid == 0) sumExp[b] = 0.0f;
}

// ---------------------------------------------------------------------------
// GRU gemms: gi = w @ Wih^T, gh = h @ Whh^T  (both 32x2304, K=768).
// 288 N-tiles total (144 per matrix), one wave computes full M=32.
// 36 blocks x 256 threads.
// ---------------------------------------------------------------------------
__global__ void gru_gemm_kernel(const _Float16* __restrict__ w16,
                                const _Float16* __restrict__ h16,
                                const _Float16* __restrict__ wih16,
                                const _Float16* __restrict__ whh16,
                                float* __restrict__ gi,
                                float* __restrict__ gh) {
    int tile = (blockIdx.x * blockDim.x + threadIdx.x) >> 5;
    const _Float16 *A, *Bm;
    float* out;
    if (tile < 144) { A = w16; Bm = wih16; out = gi; }
    else            { tile -= 144; A = h16; Bm = whh16; out = gh; }
    int n0 = tile * 16;
    int lane = threadIdx.x & 31, half = lane >> 4, l15 = lane & 15;

    const _Float16* ap0 = A  + (size_t)l15 * H_ + half * 8;
    const _Float16* ap1 = ap0 + (size_t)16 * H_;
    const _Float16* bp  = Bm + (size_t)(n0 + l15) * H_ + half * 8;
    v8f c0, c1;
    wmma_k768_m32(ap0, ap1, bp, c0, c1);

    int n = n0 + l15;
#pragma unroll
    for (int r = 0; r < 8; ++r) {
        int m = r + 8 * half;
        out[(size_t)m * H3_ + n]        = c0[r];
        out[(size_t)(m + 16) * H3_ + n] = c1[r];
    }
}

// ---------------------------------------------------------------------------
// GRU gates + attention + context + p_gen.  grid=B_, block=256.
// ---------------------------------------------------------------------------
__global__ void gates_attn_kernel(const float* __restrict__ gi,
                                  const float* __restrict__ gh,
                                  const float* __restrict__ bih,
                                  const float* __restrict__ bhh,
                                  const float* __restrict__ enc,
                                  const int* __restrict__ x,
                                  const float* __restrict__ w32,
                                  const float* __restrict__ wgw,
                                  const float* __restrict__ wgb,
                                  float* __restrict__ h32,
                                  _Float16* __restrict__ h16,
                                  float* __restrict__ attn,
                                  float* __restrict__ p_gen) {
    __shared__ float hs[H_];
    __shared__ float attns[S_];
    __shared__ float red[256];
    int b = blockIdx.x, tid = threadIdx.x;
    const float* gib = gi + (size_t)b * H3_;
    const float* ghb = gh + (size_t)b * H3_;

    // ---- GRU cell ----
#pragma unroll
    for (int j = 0; j < 3; ++j) {
        int k = tid + j * 256;
        float ir = gib[k] + bih[k];
        float hr = ghb[k] + bhh[k];
        float iz = gib[H_ + k] + bih[H_ + k];
        float hz = ghb[H_ + k] + bhh[H_ + k];
        float in_ = gib[2 * H_ + k] + bih[2 * H_ + k];
        float hn = ghb[2 * H_ + k] + bhh[2 * H_ + k];
        float r = 1.0f / (1.0f + expf(-(ir + hr)));
        float z = 1.0f / (1.0f + expf(-(iz + hz)));
        float nn = tanhf(in_ + r * hn);
        float hold = h32[(size_t)b * H_ + k];
        float hnew = (1.0f - z) * nn + z * hold;
        hs[k] = hnew;
        h32[(size_t)b * H_ + k] = hnew;
        h16[(size_t)b * H_ + k] = (_Float16)hnew;
    }
    __syncthreads();

    // ---- attention scores + masked softmax over S ----
    const float* encb = enc + (size_t)b * S_ * H_;
    int s = tid;
    float e = 0.0f;
    const float* er = encb + (size_t)s * H_;
    for (int k = 0; k < H_; ++k) e += er[k] * hs[k];
    if (x[b * S_ + s] == 0) e = -1.0e9f;

    red[tid] = e; __syncthreads();
    for (int off = 128; off > 0; off >>= 1) {
        if (tid < off) red[tid] = fmaxf(red[tid], red[tid + off]);
        __syncthreads();
    }
    float mx = red[0]; __syncthreads();
    float ex = __expf(e - mx);
    red[tid] = ex; __syncthreads();
    for (int off = 128; off > 0; off >>= 1) {
        if (tid < off) red[tid] += red[tid + off];
        __syncthreads();
    }
    float a = ex / red[0]; __syncthreads();
    attns[s] = a;
    attn[b * S_ + s] = a;
    __syncthreads();

    // ---- context + p_gen dot over [w, h, ctx] ----
    float part = 0.0f;
#pragma unroll
    for (int j = 0; j < 3; ++j) {
        int k = tid + j * 256;
        float c = 0.0f;
        for (int ss = 0; ss < S_; ++ss) c += attns[ss] * encb[(size_t)ss * H_ + k];
        part += w32[(size_t)b * H_ + k] * wgw[k];
        part += hs[k] * wgw[H_ + k];
        part += c * wgw[2 * H_ + k];
    }
    red[tid] = part; __syncthreads();
    for (int off = 128; off > 0; off >>= 1) {
        if (tid < off) red[tid] += red[tid + off];
        __syncthreads();
    }
    if (tid == 0) p_gen[b] = 1.0f / (1.0f + expf(-(red[0] + wgb[0])));
}

// ---------------------------------------------------------------------------
// Vocab gemm: logits = h @ embed^T (32 x 30522, K=768); one wave per 16-wide
// N-tile computes the full M=32 (both accumulators share the B loads).
// 1908 tiles: 239 blocks x 256 threads (last block partially idle via guard
// applied only AFTER all WMMAs for valid tiles; extra waves use tile clamp).
// ---------------------------------------------------------------------------
__global__ void vocab_gemm_kernel(const _Float16* __restrict__ h16,
                                  const _Float16* __restrict__ emb16,
                                  float* __restrict__ outStep,   // d_out + (u*ML+t)*V
                                  float* __restrict__ sumExp,
                                  long strideB) {               // NU*ML*V
    int tile = (blockIdx.x * blockDim.x + threadIdx.x) >> 5;
    int tc = tile < NTV ? tile : NTV - 1;      // clamp (branch-free, EXEC stays full)
    bool tileValid = tile < NTV;
    int n0 = tc * 16;
    int lane = threadIdx.x & 31, half = lane >> 4, l15 = lane & 15;

    int n = n0 + l15;
    int nc = n < V_ ? n : V_ - 1;              // column clamp
    const _Float16* ap0 = h16 + (size_t)l15 * H_ + half * 8;
    const _Float16* ap1 = ap0 + (size_t)16 * H_;
    const _Float16* bp  = emb16 + (size_t)nc * H_ + half * 8;
    v8f c0, c1;
    wmma_k768_m32(ap0, ap1, bp, c0, c1);

    bool valid = (n < V_) && tileValid;
#pragma unroll
    for (int r = 0; r < 8; ++r) {
        int b0 = r + 8 * half;                 // rows 0..15
        float e0 = valid ? __expf(c0[r]) : 0.0f;
        float e1 = valid ? __expf(c1[r]) : 0.0f;
        if (valid) {
            outStep[(size_t)b0 * strideB + n]        = e0;
            outStep[(size_t)(b0 + 16) * strideB + n] = e1;
        }
        float v0 = e0, v1 = e1;                // reduce across 16 lanes of this half
        v0 += __shfl_xor(v0, 1, 32);  v1 += __shfl_xor(v1, 1, 32);
        v0 += __shfl_xor(v0, 2, 32);  v1 += __shfl_xor(v1, 2, 32);
        v0 += __shfl_xor(v0, 4, 32);  v1 += __shfl_xor(v1, 4, 32);
        v0 += __shfl_xor(v0, 8, 32);  v1 += __shfl_xor(v1, 8, 32);
        if (l15 == 0) {
            atomicAdd(&sumExp[b0], v0);
            atomicAdd(&sumExp[b0 + 16], v1);
        }
    }
}

// ---------------------------------------------------------------------------
// Finalize: out *= p_gen / sumExp
// ---------------------------------------------------------------------------
__global__ void finalize_kernel(float* __restrict__ outStep,
                                const float* __restrict__ p_gen,
                                const float* __restrict__ sumExp,
                                long strideB) {
    int idx = blockIdx.x * 256 + threadIdx.x;
    if (idx >= B_ * V_) return;
    int b = idx / V_;
    int v = idx - b * V_;
    float* p = outStep + (size_t)b * strideB + v;
    *p = (*p) * p_gen[b] / sumExp[b];
}

// ---------------------------------------------------------------------------
// Pointer scatter: out[b, x[b,s]] += (1 - p_gen[b]) * attn[b,s]
// ---------------------------------------------------------------------------
__global__ void scatter_kernel(float* __restrict__ outStep,
                               const int* __restrict__ x,
                               const float* __restrict__ attn,
                               const float* __restrict__ p_gen,
                               long strideB) {
    int b = blockIdx.x, s = threadIdx.x;
    int v = x[b * S_ + s];
    atomicAdd(outStep + (size_t)b * strideB + v,
              (1.0f - p_gen[b]) * attn[b * S_ + s]);
}

// ---------------------------------------------------------------------------
extern "C" void kernel_launch(void* const* d_in, const int* in_sizes, int n_in,
                              void* d_out, int out_size, void* d_ws, size_t ws_size,
                              hipStream_t stream) {
    const int*   x       = (const int*)  d_in[0];
    const float* dec_in  = (const float*)d_in[1];
    const float* enc     = (const float*)d_in[2];
    const float* hidden  = (const float*)d_in[3];
    const int*   teacher = (const int*)  d_in[5];
    const float* embed   = (const float*)d_in[6];
    const float* wih     = (const float*)d_in[7];
    const float* whh     = (const float*)d_in[8];
    const float* bih     = (const float*)d_in[9];
    const float* bhh     = (const float*)d_in[10];
    const float* wgw     = (const float*)d_in[11];
    const float* wgb     = (const float*)d_in[12];
    float* out = (float*)d_out;

    // workspace carve-out
    char* ws = (char*)d_ws;
    size_t off = 0;
    auto carve = [&](size_t bytes) -> void* {
        void* p = ws + off;
        off = (off + bytes + 255) & ~(size_t)255;
        return p;
    };
    _Float16* emb16  = (_Float16*)carve((size_t)V_ * H_ * 2);
    _Float16* wih16  = (_Float16*)carve((size_t)H3_ * H_ * 2);
    _Float16* whh16  = (_Float16*)carve((size_t)H3_ * H_ * 2);
    float*    h32    = (float*)   carve((size_t)B_ * H_ * 4);
    _Float16* h16    = (_Float16*)carve((size_t)B_ * H_ * 2);
    float*    w32    = (float*)   carve((size_t)B_ * H_ * 4);
    _Float16* w16    = (_Float16*)carve((size_t)B_ * H_ * 2);
    float*    gi     = (float*)   carve((size_t)B_ * H3_ * 4);
    float*    gh     = (float*)   carve((size_t)B_ * H3_ * 4);
    float*    attn   = (float*)   carve((size_t)B_ * S_ * 4);
    float*    p_gen  = (float*)   carve(256);
    float*    sumExp = (float*)   carve(256);

    // one-time (per launch) conversions
    {
        long n = (long)V_ * H_;
        cvt_f16_kernel<<<(int)((n + 255) / 256), 256, 0, stream>>>(embed, emb16, n);
        long m = (long)H3_ * H_;
        cvt_f16_kernel<<<(int)((m + 255) / 256), 256, 0, stream>>>(wih, wih16, m);
        cvt_f16_kernel<<<(int)((m + 255) / 256), 256, 0, stream>>>(whh, whh16, m);
        hipMemcpyAsync(h32, hidden, (size_t)B_ * H_ * 4, hipMemcpyDeviceToDevice, stream);
    }

    const long strideB = (long)NU_ * ML_ * V_;
    const int vocabBlocks = (NTV * 32 + 255) / 256;   // 1908 waves -> 239 blocks
    for (int step = 0; step < NSTEP; ++step) {
        int u = step / ML_, t = step % ML_;
        float* outStep = out + (size_t)(u * ML_ + t) * V_;

        prep_kernel<<<B_, 256, 0, stream>>>(dec_in, teacher, embed, h32, h16,
                                            w32, w16, sumExp, u, t);
        gru_gemm_kernel<<<36, 256, 0, stream>>>(w16, h16, wih16, whh16, gi, gh);
        gates_attn_kernel<<<B_, 256, 0, stream>>>(gi, gh, bih, bhh, enc, x, w32,
                                                  wgw, wgb, h32, h16, attn, p_gen);
        vocab_gemm_kernel<<<vocabBlocks, 256, 0, stream>>>(h16, emb16, outStep,
                                                           sumExp, strideB);
        finalize_kernel<<<(B_ * V_ + 255) / 256, 256, 0, stream>>>(outStep, p_gen,
                                                                   sumExp, strideB);
        scatter_kernel<<<B_, S_, 0, stream>>>(outStep, x, attn, p_gen, strideB);
    }
}
